// MultiHeadAttention_46213848105567
// MI455X (gfx1250) — matrix operand order repaired and checked by reading, not compiled, not run on hardware
//
#include <hip/hip_runtime.h>
#include <hip/hip_bf16.h>
#include <math.h>

// MI455X / gfx1250: wave32, WMMA bf16 16x16x32 (f32 acc).
// Compute-bound MHA (~69 GFLOP vs ~3us of HBM traffic at 23.3 TB/s), so:
//  - everything runs on the bf16 WMMA pipe (8x the K-rate of f32 WMMA),
//  - global->LDS staging uses gfx1250 async-to-LDS copies (ASYNCcnt) with
//    double-buffered tiles so staging overlaps the matrix pipe,
//  - all B-operand tiles are staged transposed in LDS so every fragment
//    load is a contiguous ds_load_b128 pair (no strided u16 storms).

typedef __attribute__((ext_vector_type(16))) __bf16 v16bf;
typedef __attribute__((ext_vector_type(8)))  float  v8f;
typedef __attribute__((ext_vector_type(4)))  int    v4i_;

#define B_  2
#define LQ_ 2048
#define LM_ 2048
#define D_  1024
#define H_  16
#define DH_ 64

// ---- async global->LDS copy (guarded: falls back to plain LDS stores) ----
#if defined(__HIP_DEVICE_COMPILE__) && defined(__has_builtin)
#if __has_builtin(__builtin_amdgcn_global_load_async_to_lds_b128) && \
    __has_builtin(__builtin_amdgcn_s_wait_asynccnt)
#define USE_ASYNC_LDS 1
#endif
#endif
#ifndef USE_ASYNC_LDS
#define USE_ASYNC_LDS 0
#endif

typedef __attribute__((address_space(1))) v4i_ gv4i;  // global int4
typedef __attribute__((address_space(3))) v4i_ lv4i;  // LDS int4

__device__ static inline void cp_b128(const __bf16* gsrc, __bf16* ldst) {
#if USE_ASYNC_LDS
  __builtin_amdgcn_global_load_async_to_lds_b128((gv4i*)gsrc, (lv4i*)ldst, 0, 0);
#else
  *(uint4*)ldst = *(const uint4*)gsrc;
#endif
}

__device__ static inline void cp_wait_all() {
#if USE_ASYNC_LDS
  __builtin_amdgcn_s_wait_asynccnt(0);
#endif
}

__device__ static inline __bf16 f2bf(float f) {
  unsigned int u = __builtin_bit_cast(unsigned int, f);
  u += 0x7FFFu + ((u >> 16) & 1u);  // round-to-nearest-even
  unsigned short h = (unsigned short)(u >> 16);
  return __builtin_bit_cast(__bf16, h);
}

__device__ static inline v8f wmma_bf16(v16bf a, v16bf b, v8f c) {
  return __builtin_amdgcn_wmma_f32_16x16x32_bf16(
      false, a, false, b, (short)0, c, false, false);
}

// A fragment: 16x32 bf16 (MxK), row-major src, leading dim ld (elements).
__device__ static inline v16bf frag_a(const __bf16* src, int ld) {
  const int lane = threadIdx.x & 31;
  const int m = lane & 15, half = lane >> 4;
  const __bf16* row = src + (size_t)m * ld + half * 8;
  v16bf a;
#pragma unroll
  for (int e = 0; e < 8; ++e) a[e] = row[e];
#pragma unroll
  for (int e = 0; e < 8; ++e) a[8 + e] = row[16 + e];
  return a;
}

// B fragment from transposed storage: src holds Bt[n][k]; B[k][n] = src[n*ld + k].
// Per-lane reads are contiguous -> lowers to ds_load_b128 pairs.
__device__ static inline v16bf frag_b_t(const __bf16* src, int ld) {
  const int lane = threadIdx.x & 31;
  const int n = lane & 15, half = lane >> 4;
  const __bf16* row = src + (size_t)n * ld + half * 8;
  v16bf b;
#pragma unroll
  for (int e = 0; e < 8; ++e) b[e] = row[e];
#pragma unroll
  for (int e = 0; e < 8; ++e) b[8 + e] = row[16 + e];
  return b;
}

__global__ void cvt_f32_bf16(const float* __restrict__ in, __bf16* __restrict__ out, int n) {
  int i = blockIdx.x * blockDim.x + threadIdx.x;
  int stride = gridDim.x * blockDim.x;
  for (; i < n; i += stride) out[i] = f2bf(in[i]);
}

// C = (A[MxK] * W[KxN] + bias[N]) * scale
// Block tile 128x128, 8 waves each owning 32x64 (2 A-frags x 4 B-frags = 8 WMMA/k-step).
// A staged row-major (async), W staged transposed (manual scatter), double-buffered.
template <bool OUT_F32>
__global__ __launch_bounds__(256) void gemm_bf16(
    const __bf16* __restrict__ A, const __bf16* __restrict__ W,
    const float* __restrict__ bias, void* __restrict__ Cv,
    int M, int N, int K, float scale) {
  __shared__ __bf16 sA[2][128][40];   // 128 M-rows x 32 K (+8 pad)
  __shared__ __bf16 sBt[2][128][40];  // 128 N-rows x 32 K (+8 pad), transposed W
  const int tid = threadIdx.x;
  const int wv = tid >> 5;
  const int wrow = (wv >> 1) * 32;
  const int wcol = (wv & 1) * 64;
  const int row0 = blockIdx.y * 128;
  const int n0 = blockIdx.x * 128;

  v8f c[2][4] = {};

  auto stage = [&](int buf, int k0) {
    {  // A: 128x32, each thread 16 elts = 2x b128 (async path)
      int r = tid >> 1, cb = (tid & 1) * 16;
      const __bf16* g = A + (size_t)(row0 + r) * K + k0 + cb;
      cp_b128(g, &sA[buf][r][cb]);
      cp_b128(g + 8, &sA[buf][r][cb + 8]);
    }
    {  // W: 32x128 read row-major, scattered transposed into sBt[n][k]
      int rb = tid >> 3, cbb = (tid & 7) * 16;
      const uint4* g4 = (const uint4*)(W + (size_t)(k0 + rb) * N + n0 + cbb);
      uint4 raw[2];
      raw[0] = g4[0];
      raw[1] = g4[1];
      const __bf16* vals = (const __bf16*)raw;
#pragma unroll
      for (int j = 0; j < 16; ++j) sBt[buf][cbb + j][rb] = vals[j];
    }
  };

  stage(0, 0);
  cp_wait_all();
  __syncthreads();

  int buf = 0;
  for (int k0 = 0; k0 < K; k0 += 32) {
    if (k0 + 32 < K) stage(buf ^ 1, k0 + 32);

    v16bf a0 = frag_a(&sA[buf][wrow][0], 40);
    v16bf a1 = frag_a(&sA[buf][wrow + 16][0], 40);
#pragma unroll
    for (int nt = 0; nt < 4; ++nt) {
      v16bf b = frag_b_t(&sBt[buf][wcol + nt * 16][0], 40);
      c[0][nt] = wmma_bf16(a0, b, c[0][nt]);
      c[1][nt] = wmma_bf16(a1, b, c[1][nt]);
    }
    cp_wait_all();
    __syncthreads();
    buf ^= 1;
  }

  const int lane = tid & 31, n = lane & 15, half = lane >> 4;
#pragma unroll
  for (int mi = 0; mi < 2; ++mi) {
#pragma unroll
    for (int nt = 0; nt < 4; ++nt) {
      int col = n0 + wcol + nt * 16 + n;
      float bv = bias[col];
#pragma unroll
      for (int r = 0; r < 8; ++r) {
        int row = row0 + wrow + mi * 16 + r + 8 * half;  // C layout: M = r + 8*(lane>>4)
        float val = (c[mi][nt][r] + bv) * scale;
        if constexpr (OUT_F32)
          ((float*)Cv)[(size_t)row * N + col] = val;
        else
          ((__bf16*)Cv)[(size_t)row * N + col] = f2bf(val);
      }
    }
  }
}

// Flash attention: block = 4 waves = 64 q rows per (b,h).
// K staged async (row-major, used as Bt for Q*K^T); V staged transposed for P*V.
__global__ __launch_bounds__(128) void attn_kernel(
    const __bf16* __restrict__ Q, const __bf16* __restrict__ Km,
    const __bf16* __restrict__ Vm, const float* __restrict__ bias,
    __bf16* __restrict__ O) {
  __shared__ __bf16 sQ[64][72];
  __shared__ __bf16 sK[2][64][72];   // [t][dh]
  __shared__ __bf16 sVt[2][64][72];  // [dh][t] (transposed)
  __shared__ __bf16 sP[4][16][72];   // per-wave P tile (16 q x 64 t)

  const int tid = threadIdx.x;
  const int wv = tid >> 5;
  const int lane = tid & 31, n = lane & 15, half = lane >> 4;
  const int q0 = blockIdx.x * 64;
  const int h = blockIdx.y;
  const int bz = blockIdx.z;

  const size_t qbase = ((size_t)bz * LQ_ + q0) * D_ + h * DH_;
  const size_t kvbase = ((size_t)bz * LM_) * D_ + h * DH_;

  auto stageKV = [&](int bufi, int t) {
    int r = tid >> 1, cb = (tid & 1) * 32;
    size_t gofs = kvbase + (size_t)(t + r) * D_ + cb;
    const __bf16* gk = Km + gofs;
#pragma unroll
    for (int i = 0; i < 4; ++i) cp_b128(gk + i * 8, &sK[bufi][r][cb + i * 8]);
    // V transposed: read row (t, dh contiguous), scatter into sVt[dh][t]
    const uint4* gv4 = (const uint4*)(Vm + gofs);
    uint4 raw[4];
#pragma unroll
    for (int i = 0; i < 4; ++i) raw[i] = gv4[i];
    const __bf16* vals = (const __bf16*)raw;
#pragma unroll
    for (int j = 0; j < 32; ++j) sVt[bufi][cb + j][r] = vals[j];
  };

  {  // Q strip 64x64 (async)
    int r = tid >> 1, cb = (tid & 1) * 32;
    const __bf16* g = Q + qbase + (size_t)r * D_ + cb;
#pragma unroll
    for (int i = 0; i < 4; ++i) cp_b128(g + i * 8, &sQ[r][cb + i * 8]);
  }
  stageKV(0, 0);
  cp_wait_all();
  __syncthreads();

  const v16bf aq0 = frag_a(&sQ[wv * 16][0], 72);
  const v16bf aq1 = frag_a(&sQ[wv * 16][32], 72);

  v8f o0 = {}, o1 = {}, o2 = {}, o3 = {};
  float mrun[8], lrun[8];
#pragma unroll
  for (int r = 0; r < 8; ++r) { mrun[r] = -1e30f; lrun[r] = 0.0f; }

  int buf = 0;
  for (int t0 = 0; t0 < LM_; t0 += 64) {
    if (t0 + 64 < LM_) stageKV(buf ^ 1, t0 + 64);

    // S = Q * K^T (16 x 64 per wave) + bias(t)
    v8f s[4];
#pragma unroll
    for (int nt = 0; nt < 4; ++nt) {
      v8f acc = {};
      acc = wmma_bf16(aq0, frag_b_t(&sK[buf][nt * 16][0], 72), acc);
      acc = wmma_bf16(aq1, frag_b_t(&sK[buf][nt * 16][32], 72), acc);
      float bv = bias[(size_t)bz * LM_ + t0 + nt * 16 + n];
#pragma unroll
      for (int r = 0; r < 8; ++r) acc[r] += bv;
      s[nt] = acc;
    }

    // online softmax: rows map to (VGPR slot, lane-half); reduce within 16-lane halves
    float mnew[8], corr[8];
#pragma unroll
    for (int r = 0; r < 8; ++r) {
      float mx = fmaxf(fmaxf(s[0][r], s[1][r]), fmaxf(s[2][r], s[3][r]));
      mx = fmaxf(mx, __shfl_xor(mx, 1, 32));
      mx = fmaxf(mx, __shfl_xor(mx, 2, 32));
      mx = fmaxf(mx, __shfl_xor(mx, 4, 32));
      mx = fmaxf(mx, __shfl_xor(mx, 8, 32));
      float mn = fmaxf(mrun[r], mx);
      mnew[r] = mn;
      corr[r] = __expf(mrun[r] - mn);
    }
#pragma unroll
    for (int r = 0; r < 8; ++r) {
      float rs = 0.0f;
#pragma unroll
      for (int nt = 0; nt < 4; ++nt) {
        float p = __expf(s[nt][r] - mnew[r]);
        rs += p;
        sP[wv][r + 8 * half][nt * 16 + n] = f2bf(p);
      }
      rs += __shfl_xor(rs, 1, 32);
      rs += __shfl_xor(rs, 2, 32);
      rs += __shfl_xor(rs, 4, 32);
      rs += __shfl_xor(rs, 8, 32);
      lrun[r] = lrun[r] * corr[r] + rs;
      mrun[r] = mnew[r];
      o0[r] *= corr[r]; o1[r] *= corr[r]; o2[r] *= corr[r]; o3[r] *= corr[r];
    }

    // O += P * V (DS ops in-order per wave: sP readback is safe wave-locally)
    const v16bf ap0 = frag_a(&sP[wv][0][0], 72);
    const v16bf ap1 = frag_a(&sP[wv][0][32], 72);
    o0 = wmma_bf16(ap0, frag_b_t(&sVt[buf][0][0], 72), o0);
    o0 = wmma_bf16(ap1, frag_b_t(&sVt[buf][0][32], 72), o0);
    o1 = wmma_bf16(ap0, frag_b_t(&sVt[buf][16][0], 72), o1);
    o1 = wmma_bf16(ap1, frag_b_t(&sVt[buf][16][32], 72), o1);
    o2 = wmma_bf16(ap0, frag_b_t(&sVt[buf][32][0], 72), o2);
    o2 = wmma_bf16(ap1, frag_b_t(&sVt[buf][32][32], 72), o2);
    o3 = wmma_bf16(ap0, frag_b_t(&sVt[buf][48][0], 72), o3);
    o3 = wmma_bf16(ap1, frag_b_t(&sVt[buf][48][32], 72), o3);

    cp_wait_all();
    __syncthreads();
    buf ^= 1;
  }

#pragma unroll
  for (int r = 0; r < 8; ++r) {
    float inv = 1.0f / lrun[r];
    size_t row = (size_t)bz * LQ_ + q0 + wv * 16 + r + 8 * half;
    size_t base = row * D_ + h * DH_ + n;
    O[base + 0]  = f2bf(o0[r] * inv);
    O[base + 16] = f2bf(o1[r] * inv);
    O[base + 32] = f2bf(o2[r] * inv);
    O[base + 48] = f2bf(o3[r] * inv);
  }
}

extern "C" void kernel_launch(void* const* d_in, const int* in_sizes, int n_in,
                              void* d_out, int out_size, void* d_ws, size_t ws_size,
                              hipStream_t stream) {
  (void)in_sizes; (void)n_in; (void)out_size; (void)ws_size;
  const float* query  = (const float*)d_in[0];
  const float* memory = (const float*)d_in[1];
  const float* mbias  = (const float*)d_in[2];
  const float* Wq = (const float*)d_in[3];
  const float* bq = (const float*)d_in[4];
  const float* Wk = (const float*)d_in[5];
  const float* bk = (const float*)d_in[6];
  const float* Wv = (const float*)d_in[7];
  const float* bv = (const float*)d_in[8];
  const float* Wo = (const float*)d_in[9];
  const float* bo = (const float*)d_in[10];

  const int ACT = B_ * LQ_ * D_;  // 4M elements
  const int WSZ = D_ * D_;        // 1M elements
  // Workspace (bf16): qry, mem, 4 weights, q, k, v, attn = 56 MB total.
  __bf16* p = (__bf16*)d_ws;
  __bf16* qryb = p; p += ACT;
  __bf16* memb = p; p += ACT;
  __bf16* wqb  = p; p += WSZ;
  __bf16* wkb  = p; p += WSZ;
  __bf16* wvb  = p; p += WSZ;
  __bf16* wob  = p; p += WSZ;
  __bf16* qb   = p; p += ACT;
  __bf16* kb   = p; p += ACT;
  __bf16* vb   = p; p += ACT;
  __bf16* attnb = p;

  cvt_f32_bf16<<<2048, 256, 0, stream>>>(query,  qryb, ACT);
  cvt_f32_bf16<<<2048, 256, 0, stream>>>(memory, memb, ACT);
  cvt_f32_bf16<<<1024, 256, 0, stream>>>(Wq, wqb, WSZ);
  cvt_f32_bf16<<<1024, 256, 0, stream>>>(Wk, wkb, WSZ);
  cvt_f32_bf16<<<1024, 256, 0, stream>>>(Wv, wvb, WSZ);
  cvt_f32_bf16<<<1024, 256, 0, stream>>>(Wo, wob, WSZ);

  const int M = B_ * LQ_;          // 4096
  dim3 ggrid(D_ / 128, M / 128);   // (8, 32)
  gemm_bf16<false><<<ggrid, 256, 0, stream>>>(qryb, wqb, bq, qb, M, D_, D_, 0.125f);  // * DH^-0.5
  gemm_bf16<false><<<ggrid, 256, 0, stream>>>(memb, wkb, bk, kb, M, D_, D_, 1.0f);
  gemm_bf16<false><<<ggrid, 256, 0, stream>>>(memb, wvb, bv, vb, M, D_, D_, 1.0f);

  attn_kernel<<<dim3(LQ_ / 64, H_, B_), 128, 0, stream>>>(qb, kb, vb, mbias, attnb);

  gemm_bf16<true><<<ggrid, 256, 0, stream>>>(attnb, wob, bo, d_out, M, D_, D_, 1.0f);
}
